// FullModel_65180423684893
// MI455X (gfx1250) — matrix operand order repaired and checked by reading, not compiled, not run on hardware
//
#include <hip/hip_runtime.h>
#include <math.h>

// ---------------------------------------------------------------------------
// Types
// ---------------------------------------------------------------------------
typedef unsigned short u16t;
typedef __attribute__((ext_vector_type(16))) __bf16 bf16x16;
typedef __attribute__((ext_vector_type(8)))  float  f32x8;

__device__ __forceinline__ u16t f32_to_bf16_rne(float x) {
  union { float f; unsigned u; } c; c.f = x;
  unsigned u = c.u;
  u += 0x7fffu + ((u >> 16) & 1u);   // round to nearest even
  return (u16t)(u >> 16);
}

// gfx1250 async global->LDS copy (VGLOBAL encoding, GVS mode:
// mem addr = SADDR(64) + VADDR(32) ; LDS dest byte addr in VDST VGPR).
// Tracked by ASYNCcnt; drained with s_wait_asynccnt.
__device__ __forceinline__ void async_load_b128(unsigned lds_byte, unsigned gl_byte,
                                                const void* base) {
  asm volatile("global_load_async_to_lds_b128 %0, %1, %2"
               :
               : "v"(lds_byte), "v"(gl_byte), "s"(base)
               : "memory");
}
__device__ __forceinline__ void wait_async0() {
  asm volatile("s_wait_asynccnt 0" ::: "memory");
}

// ---------------------------------------------------------------------------
// fp32 -> bf16 conversion with optional K padding (pad columns are zero)
// ---------------------------------------------------------------------------
__global__ void cvt_pad_kernel(const float* __restrict__ src, u16t* __restrict__ dst,
                               int rows, int srcK, int dstK) {
  long long idx = (long long)blockIdx.x * blockDim.x + threadIdx.x;
  long long total = (long long)rows * dstK;
  if (idx >= total) return;
  int r = (int)(idx / dstK);
  int c = (int)(idx - (long long)r * dstK);
  float v = (c < srcK) ? src[(long long)r * srcK + c] : 0.0f;
  dst[idx] = f32_to_bf16_rne(v);
}

__global__ void cvt_pack2_kernel(const float* __restrict__ w0, const float* __restrict__ w1,
                                 u16t* __restrict__ dst, int rowsEach, int srcK, int dstK) {
  long long idx = (long long)blockIdx.x * blockDim.x + threadIdx.x;
  long long total = (long long)(2 * rowsEach) * dstK;
  if (idx >= total) return;
  int r = (int)(idx / dstK);
  int c = (int)(idx - (long long)r * dstK);
  const float* s = (r < rowsEach) ? (w0 + (long long)r * srcK)
                                  : (w1 + (long long)(r - rowsEach) * srcK);
  float v = (c < srcK) ? s[c] : 0.0f;
  dst[idx] = f32_to_bf16_rne(v);
}

// ---------------------------------------------------------------------------
// WMMA GEMM:  C[M,1024] = relu( A[M,Kpad](bf16) * W[1024,Kpad](bf16)^T )
// Block: 256 threads (8 waves). Block tile 256(M) x 64(N), K step 32.
// Wave w computes rows [bm*256 + 32w, +32) x 64 cols: 2 A-frags, 4 B-frags,
// 8 WMMAs per K step (12 ds_load_b128 feeds, B-frags register-reused).
// Global->LDS tile fill uses global_load_async_to_lds_b128 (ASYNCcnt),
// double buffered; padded LDS row stride 40 elems (80B, 16B-aligned rows).
// ---------------------------------------------------------------------------
#define TILE_M 256
#define TILE_N 64
#define KSTEP  32
#define LDT    40

// element offsets inside the single LDS arena
#define A_ELEMS   (TILE_M * LDT)              // 10240 per buffer
#define B_ELEMS   (TILE_N * LDT)              // 2560 per buffer
#define B_BASE    (2 * A_ELEMS)               // after the two A buffers
#define SMEM_TOT  (2 * A_ELEMS + 2 * B_ELEMS) // 25600 u16 = 50 KB

__global__ __launch_bounds__(256)
void gemm_relu_bf16_kernel(const u16t* __restrict__ A, const u16t* __restrict__ W,
                           float* __restrict__ C, int M, int Kpad) {
  constexpr int N = 1024;
  __shared__ alignas(16) u16t smem[SMEM_TOT];

  const int tid  = threadIdx.x;
  const int lane = tid & 31;
  const int wave = tid >> 5;
  const int nBlk = N / TILE_N;                 // 16
  const int bN   = blockIdx.x % nBlk;
  const int bM   = blockIdx.x / nBlk;

  const int rowBase = bM * TILE_M;
  const int colBase = bN * TILE_N;

  // ---- async staging assignments (16B chunks of 8 bf16) -------------------
  // A tile: 256 rows x 4 chunks = 1024 chunks -> 4 per thread
  // W tile:  64 rows x 4 chunks =  256 chunks -> 1 per thread
  const int ch    = tid & 3;                   // chunk-in-row (same for all)
  const int chB   = ch * 16;                   // chunk byte offset in LDS row
  const int aR0   = tid >> 2;                  // rows for the 4 A chunks
  const int aR1   = (tid + 256) >> 2;
  const int aR2   = (tid + 512) >> 2;
  const int aR3   = (tid + 768) >> 2;
  const int wR    = tid >> 2;                  // 0..63

  // per-thread global byte offsets at K=0
  const unsigned gA0 = (unsigned)(((rowBase + aR0) * Kpad + ch * 8) * 2);
  const unsigned gA1 = (unsigned)(((rowBase + aR1) * Kpad + ch * 8) * 2);
  const unsigned gA2 = (unsigned)(((rowBase + aR2) * Kpad + ch * 8) * 2);
  const unsigned gA3 = (unsigned)(((rowBase + aR3) * Kpad + ch * 8) * 2);
  const unsigned gW  = (unsigned)(((colBase + wR) * Kpad + ch * 8) * 2);

  // LDS byte addresses (flat LDS addr carries DS offset in low 32 bits)
  const unsigned ldsBase = (unsigned)(uintptr_t)&smem[0];
  const unsigned lA0 = ldsBase + (unsigned)(aR0 * (LDT * 2) + chB);
  const unsigned lA1 = ldsBase + (unsigned)(aR1 * (LDT * 2) + chB);
  const unsigned lA2 = ldsBase + (unsigned)(aR2 * (LDT * 2) + chB);
  const unsigned lA3 = ldsBase + (unsigned)(aR3 * (LDT * 2) + chB);
  const unsigned lW  = ldsBase + (unsigned)(B_BASE * 2 + wR * (LDT * 2) + chB);
  const unsigned abufB = A_ELEMS * 2;          // byte stride between A buffers
  const unsigned bbufB = B_ELEMS * 2;          // byte stride between B buffers

  const int nsteps = Kpad / KSTEP;

  // prologue: async-fill tile 0 into buffer 0
  async_load_b128(lA0, gA0, A);
  async_load_b128(lA1, gA1, A);
  async_load_b128(lA2, gA2, A);
  async_load_b128(lA3, gA3, A);
  async_load_b128(lW,  gW,  W);
  wait_async0();
  __syncthreads();

  f32x8 acc[2][4] = {};

  const int mLane = lane & 15;
  const int hi    = lane >> 4;                 // lane half selects K sub-ranges
  const int row0  = wave * 32 + mLane;         // A frag row, group 0
  const int row1  = row0 + 16;                 // A frag row, group 1

  for (int s = 0; s < nsteps; ++s) {
    const int cur = s & 1;
    const bool hasNext = (s + 1) < nsteps;
    if (hasNext) {
      const int nxt = cur ^ 1;
      const unsigned kB = (unsigned)((s + 1) * KSTEP * 2);   // byte advance
      const unsigned la = (unsigned)(nxt ? abufB : 0);
      const unsigned lb = (unsigned)(nxt ? bbufB : 0);
      async_load_b128(lA0 + la, gA0 + kB, A);
      async_load_b128(lA1 + la, gA1 + kB, A);
      async_load_b128(lA2 + la, gA2 + kB, A);
      async_load_b128(lA3 + la, gA3 + kB, A);
      async_load_b128(lW  + lb, gW  + kB, W);
    }

    // ---- load ALL fragments first, then issue the 8 WMMAs ----
    const u16t* aT = smem + cur * A_ELEMS;
    const u16t* bT = smem + B_BASE + cur * B_ELEMS;

    union Frag { uint4 q[2]; bf16x16 v; };
    Frag a0, a1, b[4];
    {
      // A frag: lane<16 -> K {0..7, 16..23}; lane>=16 -> K {8..15, 24..31}
      const int o0 = hi ? 8 : 0, o1 = hi ? 24 : 16;
      const u16t* p0 = aT + row0 * LDT;
      const u16t* p1 = aT + row1 * LDT;
      a0.q[0] = *(const uint4*)(p0 + o0);
      a0.q[1] = *(const uint4*)(p0 + o1);
      a1.q[0] = *(const uint4*)(p1 + o0);
      a1.q[1] = *(const uint4*)(p1 + o1);
    }
    #pragma unroll
    for (int j = 0; j < 4; ++j) {
      // B frag: lane's N col = j*16 + (lane&15); lane<16 -> K {0..7, 16..23}; lane>=16 -> K {8..15, 24..31}
      const u16t* bp = bT + (j * 16 + mLane) * LDT + (hi ? 8 : 0);
      b[j].q[0] = *(const uint4*)(bp);
      b[j].q[1] = *(const uint4*)(bp + 16);
    }
    #pragma unroll
    for (int j = 0; j < 4; ++j) {
      acc[0][j] = __builtin_amdgcn_wmma_f32_16x16x32_bf16(
          false, a0.v, false, b[j].v, (short)0, acc[0][j], false, false);
      acc[1][j] = __builtin_amdgcn_wmma_f32_16x16x32_bf16(
          false, a1.v, false, b[j].v, (short)0, acc[1][j], false, false);
    }

    if (hasNext) wait_async0();
    __syncthreads();
  }

  // D layout: VGPR r, lane L -> M = r + 8*(L>>4), N = L&15 ; fuse ReLU
  #pragma unroll
  for (int rg = 0; rg < 2; ++rg) {
    #pragma unroll
    for (int j = 0; j < 4; ++j) {
      const int col = colBase + j * 16 + mLane;
      #pragma unroll
      for (int r = 0; r < 8; ++r) {
        const int m = rowBase + wave * 32 + rg * 16 + r + 8 * hi;
        float v = acc[rg][j][r];
        C[(size_t)m * N + col] = v > 0.0f ? v : 0.0f;
      }
    }
  }
}

// ---------------------------------------------------------------------------
// Tail: per-group attention loop + head MLP, one 256-thread block per group.
// f_kv: [B*256, 1024] (cols 0..511 = f_k, 512..1023 = f_v), v_kv likewise.
// ---------------------------------------------------------------------------
__device__ __forceinline__ void block_softmax(float* vals, float* sred, int n, int t) {
  float x = (t < n) ? vals[t] : -3.0e38f;
  sred[t] = x;
  __syncthreads();
  for (int off = 128; off > 0; off >>= 1) {
    if (t < off) sred[t] = fmaxf(sred[t], sred[t + off]);
    __syncthreads();
  }
  float m = sred[0];
  __syncthreads();
  float e = (t < n) ? expf(vals[t] - m) : 0.0f;
  sred[t] = e;
  __syncthreads();
  for (int off = 128; off > 0; off >>= 1) {
    if (t < off) sred[t] += sred[t + off];
    __syncthreads();
  }
  float inv = 1.0f / sred[0];
  __syncthreads();
  if (t < n) vals[t] = e * inv;
}

__global__ __launch_bounds__(256)
void tail_kernel(const float* __restrict__ metas, const float* __restrict__ audios,
                 const float* __restrict__ Wav,
                 const float* __restrict__ Wff_f, const float* __restrict__ Wff_v,
                 const float* __restrict__ Wff_a,
                 const float* __restrict__ Wout1, const float* __restrict__ Wout2,
                 const float* __restrict__ Wout3, const float* __restrict__ Wout4,
                 const float* __restrict__ f_kv, const float* __restrict__ v_kv,
                 float* __restrict__ out) {
  const int b = blockIdx.x;
  const int t = threadIdx.x;

  __shared__ float fq[512], vq[512], aq[512], av[512];
  __shared__ float fc[512], vc[512];
  __shared__ float nf[512], nv[512], na_[512];
  __shared__ float attn[256];
  __shared__ float sred[256];
  __shared__ float feats[1539];
  __shared__ float h1[768], h2[512], h3[64];
  __shared__ float sc[2];

  const float* fk_base = f_kv + (size_t)b * 256 * 1024;
  const float* vk_base = v_kv + (size_t)b * 128 * 1024;

  // a_v = relu(audios @ Wav^T); initial a_q_p = a_v
  for (int c = t; c < 512; c += 256) {
    float s = 0.0f;
    const float* wr = Wav + (size_t)c * 128;
    const float* ar = audios + (size_t)b * 128;
    for (int k = 0; k < 128; ++k) s += ar[k] * wr[k];
    s = fmaxf(s, 0.0f);
    av[c] = s; aq[c] = s;
  }
  // initial queries: sums of value projections (softmax over (N,1) -> ones)
  for (int c = t; c < 512; c += 256) {
    float s = 0.0f;
    for (int f = 0; f < 256; ++f) s += fk_base[(size_t)f * 1024 + 512 + c];
    fq[c] = s;
    float s2 = 0.0f;
    for (int v = 0; v < 128; ++v) s2 += vk_base[(size_t)v * 1024 + 512 + c];
    vq[c] = s2;
  }
  __syncthreads();

  for (int it = 0; it < 2; ++it) {
    const float* Wf = Wff_f + (size_t)it * 512 * 512;
    const float* Wv = Wff_v + (size_t)it * 512 * 512;
    const float* Wa = Wff_a + (size_t)it * 512 * 512;

    // frame attention: logits = (f_k . v_q_p) / 512
    {
      float s = 0.0f;
      const float* row = fk_base + (size_t)t * 1024;
      for (int k = 0; k < 512; ++k) s += row[k] * vq[k];
      attn[t] = s * (1.0f / 512.0f);
    }
    __syncthreads();
    block_softmax(attn, sred, 256, t);
    __syncthreads();
    for (int c = t; c < 512; c += 256) {
      float s = 0.0f;
      for (int f = 0; f < 256; ++f) s += attn[f] * fk_base[(size_t)f * 1024 + 512 + c];
      fc[c] = s;
    }
    __syncthreads();

    // vec attention: logits = (v_k . v_q_p) / 512
    if (t < 128) {
      float s = 0.0f;
      const float* row = vk_base + (size_t)t * 1024;
      for (int k = 0; k < 512; ++k) s += row[k] * vq[k];
      attn[t] = s * (1.0f / 512.0f);
    }
    __syncthreads();
    block_softmax(attn, sred, 128, t);
    __syncthreads();
    for (int c = t; c < 512; c += 256) {
      float s = 0.0f;
      for (int v = 0; v < 128; ++v) s += attn[v] * vk_base[(size_t)v * 1024 + 512 + c];
      vc[c] = s;
    }
    __syncthreads();

    // FF updates (from old states)
    for (int c = t; c < 512; c += 256) {
      float sf = fq[c], sv = vq[c], sa = aq[c];
      const float* wfr = Wf + (size_t)c * 512;
      const float* wvr = Wv + (size_t)c * 512;
      const float* war = Wa + (size_t)c * 512;
      for (int k = 0; k < 512; ++k) {
        sf += wfr[k] * fc[k];
        sv += wvr[k] * vc[k];
        sa += war[k] * av[k];
      }
      nf[c] = sf; nv[c] = sv; na_[c] = sa;
    }
    __syncthreads();
    for (int c = t; c < 512; c += 256) { fq[c] = nf[c]; vq[c] = nv[c]; aq[c] = na_[c]; }
    __syncthreads();
  }

  // normalize a_q_p: mean and unbiased std over 512
  {
    sred[t] = aq[t] + aq[t + 256];
    __syncthreads();
    for (int off = 128; off > 0; off >>= 1) {
      if (t < off) sred[t] += sred[t + off];
      __syncthreads();
    }
    if (t == 0) sc[0] = sred[0] * (1.0f / 512.0f);
    __syncthreads();
    float mean = sc[0];
    float d0 = aq[t] - mean, d1 = aq[t + 256] - mean;
    sred[t] = d0 * d0 + d1 * d1;
    __syncthreads();
    for (int off = 128; off > 0; off >>= 1) {
      if (t < off) sred[t] += sred[t + off];
      __syncthreads();
    }
    if (t == 0) sc[1] = sqrtf(sred[0] * (1.0f / 511.0f));
    __syncthreads();
    float mean2 = sc[0], inv_std = 1.0f / sc[1];
    for (int c = t; c < 512; c += 256) {
      feats[c]        = fq[c];
      feats[512 + c]  = vq[c];
      feats[1024 + c] = (aq[c] - mean2) * inv_std;
    }
    if (t == 0) {
      const float* mrow = metas + (size_t)b * 5;
      float goal = mrow[1] * (1.0f / 1500.0f);
      float dura = mrow[3] - mrow[2];
      dura = (dura > 8000.0f) ? 100.0f : dura;
      dura *= (1.0f / 100.0f);
      float create = mrow[4] * (1.0f / 5.0f);
      feats[1536] = goal; feats[1537] = create; feats[1538] = dura;
    }
  }
  __syncthreads();

  // head MLP
  for (int r = t; r < 768; r += 256) {
    float s = 0.0f;
    const float* wr = Wout1 + (size_t)r * 1539;
    for (int k = 0; k < 1539; ++k) s += wr[k] * feats[k];
    h1[r] = fmaxf(s, 0.0f);
  }
  __syncthreads();
  for (int r = t; r < 512; r += 256) {
    float s = 0.0f;
    const float* wr = Wout2 + (size_t)r * 768;
    for (int k = 0; k < 768; ++k) s += wr[k] * h1[k];
    h2[r] = fmaxf(s, 0.0f);
  }
  __syncthreads();
  if (t < 64) {
    float s = 0.0f;
    const float* wr = Wout3 + (size_t)t * 512;
    for (int k = 0; k < 512; ++k) s += wr[k] * h2[k];
    h3[t] = fmaxf(s, 0.0f);
  }
  __syncthreads();
  if (t == 0) {
    float s = 0.0f;
    for (int k = 0; k < 64; ++k) s += Wout4[k] * h3[k];
    out[b] = 1.0f / (1.0f + expf(-s));
  }
}

// ---------------------------------------------------------------------------
// Launch
// ---------------------------------------------------------------------------
extern "C" void kernel_launch(void* const* d_in, const int* in_sizes, int n_in,
                              void* d_out, int out_size, void* d_ws, size_t ws_size,
                              hipStream_t stream) {
  (void)in_sizes; (void)n_in; (void)out_size; (void)ws_size;

  const float* metas  = (const float*)d_in[0];
  const float* frames = (const float*)d_in[2];
  const float* vecs   = (const float*)d_in[4];
  const float* audios = (const float*)d_in[6];
  const float* Wfk    = (const float*)d_in[8];
  const float* Wfv    = (const float*)d_in[9];
  const float* Wvk    = (const float*)d_in[10];
  const float* Wvv    = (const float*)d_in[11];
  const float* Wav    = (const float*)d_in[12];
  const float* Wff_f  = (const float*)d_in[13];
  const float* Wff_v  = (const float*)d_in[14];
  const float* Wff_a  = (const float*)d_in[15];
  const float* Wout1  = (const float*)d_in[16];
  const float* Wout2  = (const float*)d_in[17];
  const float* Wout3  = (const float*)d_in[18];
  const float* Wout4  = (const float*)d_in[19];
  float* out = (float*)d_out;

  const int B = 64, F = 256, V = 128;
  const int MF = B * F;           // 16384
  const int MV = B * V;           // 8192
  const int KF = 608;             // 600 padded to 19*32
  const int KV = 768;             // 24*32

  char* ws = (char*)d_ws;
  size_t off = 0;
  auto carve = [&](size_t bytes) -> char* {
    char* p = ws + off;
    off = (off + bytes + 255) & ~(size_t)255;
    return p;
  };
  u16t*  fr_bf = (u16t*)carve((size_t)MF * KF * 2);
  u16t*  wf_bf = (u16t*)carve((size_t)1024 * KF * 2);
  u16t*  vc_bf = (u16t*)carve((size_t)MV * KV * 2);
  u16t*  wv_bf = (u16t*)carve((size_t)1024 * KV * 2);
  float* f_kv  = (float*)carve((size_t)MF * 1024 * 4);
  float* v_kv  = (float*)carve((size_t)MV * 1024 * 4);

  {
    long long n = (long long)MF * KF;
    cvt_pad_kernel<<<(unsigned)((n + 255) / 256), 256, 0, stream>>>(frames, fr_bf, MF, 600, KF);
  }
  {
    long long n = (long long)1024 * KF;
    cvt_pack2_kernel<<<(unsigned)((n + 255) / 256), 256, 0, stream>>>(Wfk, Wfv, wf_bf, 512, 600, KF);
  }
  {
    long long n = (long long)MV * KV;
    cvt_pad_kernel<<<(unsigned)((n + 255) / 256), 256, 0, stream>>>(vecs, vc_bf, MV, 768, KV);
  }
  {
    long long n = (long long)1024 * KV;
    cvt_pack2_kernel<<<(unsigned)((n + 255) / 256), 256, 0, stream>>>(Wvk, Wvv, wv_bf, 512, 768, KV);
  }

  gemm_relu_bf16_kernel<<<(MF / TILE_M) * (1024 / TILE_N), 256, 0, stream>>>(fr_bf, wf_bf, f_kv, MF, KF);
  gemm_relu_bf16_kernel<<<(MV / TILE_M) * (1024 / TILE_N), 256, 0, stream>>>(vc_bf, wv_bf, v_kv, MV, KV);

  tail_kernel<<<B, 256, 0, stream>>>(metas, audios, Wav, Wff_f, Wff_v, Wff_a,
                                     Wout1, Wout2, Wout3, Wout4, f_kv, v_kv, out);
}
